// NonUniformRVQ_31602369364120
// MI455X (gfx1250) — matrix operand order repaired and checked by reading, not compiled, not run on hardware
//
#include <hip/hip_runtime.h>

// ---------------------------------------------------------------------------
// CDNA5 (gfx1250) residual VQ, wave32 + v_wmma_f32_16x16x32_bf16.
// Each wave owns 32 token rows (two 16-row A tiles) so every B fragment
// (codebook, L2-resident) feeds TWO wmmas -> halves L2 traffic vs 16-row tiles.
// ---------------------------------------------------------------------------

typedef __attribute__((ext_vector_type(16))) __bf16 v16bf;
typedef __attribute__((ext_vector_type(8)))  float  v8f;
typedef __attribute__((ext_vector_type(4)))  __bf16 bf16x4;
typedef unsigned int u32x4 __attribute__((ext_vector_type(4)));
typedef float f32x4 __attribute__((ext_vector_type(4)));

#define DIMS       768
#define DIMS4      192            // DIMS / 4
#define TILE       128            // tokens per workgroup (4 waves x 32 rows)
#define BLOCKDIM   128
#define ROWSTRIDE  776            // bf16 elems per LDS row (768 + 8 pad -> bank skew)
#define RES_BYTES  (TILE * ROWSTRIDE * 2)        // 198656
#define LIDX_OFF   RES_BYTES                     // 128 ints
#define WSUM_OFF   (RES_BYTES + TILE * 4)        // 4 floats
#define LDS_BYTES  (WSUM_OFF + 16)

union ABFrag { v16bf v; u32x4 q[2]; };

// ---------------------------------------------------------------------------
// Prep: f32 codebook -> bf16 (rows padded to Kpad with zeros) + ||c||^2
// (pad rows get 1e30 so they never win the argmin). Deterministic tree reduce.
// ---------------------------------------------------------------------------
__global__ __launch_bounds__(256)
void rvq_prep(const float* __restrict__ cb, int K,
              unsigned short* __restrict__ w_u, float* __restrict__ nrm)
{
    __bf16* w = (__bf16*)w_u;
    const int r = blockIdx.x;
    const int t = threadIdx.x;
    __shared__ float red[256];
    float s = 0.f;
    __bf16* dst = w + (size_t)r * DIMS;
    if (r < K) {
        const float* src = cb + (size_t)r * DIMS;
        for (int c = t; c < DIMS; c += 256) {
            float v = src[c];
            dst[c] = (__bf16)v;
            s += v * v;
        }
    } else {
        for (int c = t; c < DIMS; c += 256) dst[c] = (__bf16)0.f;
    }
    red[t] = s;
    __syncthreads();
    for (int o = 128; o > 0; o >>= 1) {
        if (t < o) red[t] += red[t + o];
        __syncthreads();
    }
    if (t == 0) nrm[r] = (r < K) ? red[0] : 1.0e30f;
}

// ---------------------------------------------------------------------------
// Main: one workgroup = 128 tokens. Residual lives in LDS (bf16) for all 8
// layers. Scores via WMMA bf16; argmin via wave32 shfl_xor butterflies.
// ---------------------------------------------------------------------------
__global__ __launch_bounds__(BLOCKDIM)
void rvq_main(const float* __restrict__ x,
              const float* __restrict__ cbp0, const float* __restrict__ cbp1,
              const float* __restrict__ cbp2, const float* __restrict__ cbp3,
              const float* __restrict__ cbp4, const float* __restrict__ cbp5,
              const float* __restrict__ cbp6, const float* __restrict__ cbp7,
              const unsigned short* __restrict__ cbw_u,
              const float* __restrict__ cnorm,
              float* __restrict__ outq,
              float* __restrict__ outidx,
              float* __restrict__ partial)
{
    extern __shared__ char smem[];
    __bf16* res  = (__bf16*)smem;
    int*    lidx = (int*)(smem + LIDX_OFF);
    float*  wsum = (float*)(smem + WSUM_OFF);
    const __bf16* cbw = (const __bf16*)cbw_u;

    const int tid  = threadIdx.x;
    const int lane = tid & 31;
    const int wv   = tid >> 5;       // wave 0..3, owns rows 32*wv .. 32*wv+31
    const int half = lane >> 4;      // lane half (0/1) per WMMA layouts
    const int l16  = lane & 15;
    const long tile0 = (long)blockIdx.x * TILE;

    // ---- Phase A: x tile (f32, global) -> residual tile (bf16, LDS) ----
    for (int i = tid; i < TILE * DIMS4; i += BLOCKDIM) {
        const int row = i / DIMS4;
        const int d0  = (i - row * DIMS4) * 4;
        const f32x4 xv = *(const f32x4*)(x + (tile0 + row) * DIMS + d0);
        bf16x4 rb;
        rb.x = (__bf16)xv.x; rb.y = (__bf16)xv.y;
        rb.z = (__bf16)xv.z; rb.w = (__bf16)xv.w;
        *(bf16x4*)(res + row * ROWSTRIDE + d0) = rb;
    }

    const float* cbs[8] = {cbp0, cbp1, cbp2, cbp3, cbp4, cbp5, cbp6, cbp7};
    const int Kpad[8]   = {16, 16, 16, 16, 32, 64, 128, 256};
    const int rowoff[8] = {0, 16, 32, 48, 64, 96, 160, 288};

    // A fragment sources: this wave's two 16-row tiles (M = l16 within tile).
    const __bf16* arow0 = res + (wv * 32 + l16) * ROWSTRIDE;
    const __bf16* arow1 = arow0 + 16 * ROWSTRIDE;
    const int aoff = half * 8;       // ISA: A lane<16 -> K {0..7,16..23}; lane>=16 -> {8..15,24..31}

    float lsum = 0.f;                // commit-loss accumulator (sum of new-residual^2)

#pragma unroll
    for (int l = 0; l < 8; ++l) {
        __syncthreads();             // residual tile consistent

        const int kp   = Kpad[l];
        const int roff = rowoff[l];

        float bv0[8], bv1[8];
        int   bi0[8], bi1[8];
#pragma unroll
        for (int j = 0; j < 8; ++j) {
            bv0[j] = 3.0e38f; bi0[j] = 0;
            bv1[j] = 3.0e38f; bi1[j] = 0;
        }

        for (int ct = 0; ct < kp / 16; ++ct) {
            // B fragment: codeword column N = l16 of this 16-wide tile.
            // B lane<16 -> K {0..7,16..23}; lane>=16 -> {8..15,24..31} (matches A).
            const __bf16* brow =
                cbw + (size_t)(roff + ct * 16 + l16) * DIMS + half * 8;
            if (ct + 1 < kp / 16)
                __builtin_prefetch(brow + 16 * DIMS, 0, 3);   // next tile's row

            v8f acc0 = {};
            v8f acc1 = {};
#pragma unroll 4
            for (int dc = 0; dc < DIMS; dc += 32) {
                ABFrag a0, a1, b;
                b.q[0]  = *(const u32x4*)(brow + dc);
                b.q[1]  = *(const u32x4*)(brow + dc + 16);
                a0.q[0] = *(const u32x4*)(arow0 + dc + aoff);
                a0.q[1] = *(const u32x4*)(arow0 + dc + 16 + aoff);
                a1.q[0] = *(const u32x4*)(arow1 + dc + aoff);
                a1.q[1] = *(const u32x4*)(arow1 + dc + 16 + aoff);
                acc0 = __builtin_amdgcn_wmma_f32_16x16x32_bf16(
                    false, a0.v, false, b.v, (short)0, acc0, false, false);
                acc1 = __builtin_amdgcn_wmma_f32_16x16x32_bf16(
                    false, a1.v, false, b.v, (short)0, acc1, false, false);
            }
            // score = ||c||^2 - 2 * <r, c>   (argmin-equivalent to distance)
            const float cn   = cnorm[roff + ct * 16 + l16];
            const int   cand = ct * 16 + l16;
#pragma unroll
            for (int j = 0; j < 8; ++j) {
                const float s0 = __builtin_fmaf(-2.f, acc0[j], cn);
                if (s0 < bv0[j]) { bv0[j] = s0; bi0[j] = cand; }
                const float s1 = __builtin_fmaf(-2.f, acc1[j], cn);
                if (s1 < bv1[j]) { bv1[j] = s1; bi1[j] = cand; }
            }
        }

        // Argmin over the 16 columns: xor-butterfly stays inside each 16-lane
        // half (C/D layout: lanes 0-15 hold rows +j, lanes 16-31 rows +8+j).
#pragma unroll
        for (int j = 0; j < 8; ++j) {
#pragma unroll
            for (int m = 8; m >= 1; m >>= 1) {
                const float ov0 = __shfl_xor(bv0[j], m, 32);
                const int   oi0 = __shfl_xor(bi0[j], m, 32);
                if (ov0 < bv0[j] || (ov0 == bv0[j] && oi0 < bi0[j])) {
                    bv0[j] = ov0; bi0[j] = oi0;
                }
                const float ov1 = __shfl_xor(bv1[j], m, 32);
                const int   oi1 = __shfl_xor(bi1[j], m, 32);
                if (ov1 < bv1[j] || (ov1 == bv1[j] && oi1 < bi1[j])) {
                    bv1[j] = ov1; bi1[j] = oi1;
                }
            }
        }

        if (l16 == 0) {
#pragma unroll
            for (int j = 0; j < 8; ++j) {
                const int  row0 = wv * 32 + half * 8 + j;
                const int  row1 = row0 + 16;
                lidx[row0] = bi0[j];
                lidx[row1] = bi1[j];
                outidx[(tile0 + row0) * 8 + l] = (float)bi0[j];
                outidx[(tile0 + row1) * 8 + l] = (float)bi1[j];
            }
        }
        __syncthreads();             // indices ready; all WMMA reads done

        // ---- Residual update + commit accumulation (block-flat) ----
        const float* cbl = cbs[l];
        for (int i = tid; i < TILE * DIMS4; i += BLOCKDIM) {
            const int row = i / DIMS4;
            const int d0  = (i - row * DIMS4) * 4;
            const int kidx = lidx[row];
            const f32x4 cv = *(const f32x4*)(cbl + (size_t)kidx * DIMS + d0);
            __bf16* rp = res + row * ROWSTRIDE + d0;
            bf16x4 rb = *(bf16x4*)rp;
            const float n0 = (float)rb.x - cv.x;
            const float n1 = (float)rb.y - cv.y;
            const float n2 = (float)rb.z - cv.z;
            const float n3 = (float)rb.w - cv.w;
            lsum += n0 * n0 + n1 * n1 + n2 * n2 + n3 * n3;   // (q - r)^2 == new_r^2
            rb.x = (__bf16)n0; rb.y = (__bf16)n1;
            rb.z = (__bf16)n2; rb.w = (__bf16)n3;
            *(bf16x4*)rp = rb;
        }
    }

    // ---- Commit-loss partial: deterministic wave shuffle + LDS reduce ----
#pragma unroll
    for (int m = 16; m >= 1; m >>= 1) lsum += __shfl_xor(lsum, m, 32);
    if (lane == 0) wsum[wv] = lsum;
    __syncthreads();                 // also makes final residual visible below
    if (tid == 0)
        partial[blockIdx.x] = (wsum[0] + wsum[1]) + (wsum[2] + wsum[3]);

    // ---- Phase C: quantized = x - final residual ----
    for (int i = tid; i < TILE * DIMS4; i += BLOCKDIM) {
        const int row = i / DIMS4;
        const int d0  = (i - row * DIMS4) * 4;
        const long tok = tile0 + row;
        const f32x4 xv = *(const f32x4*)(x + tok * DIMS + d0);
        const bf16x4 rb = *(const bf16x4*)(res + row * ROWSTRIDE + d0);
        f32x4 q;
        q.x = xv.x - (float)rb.x; q.y = xv.y - (float)rb.y;
        q.z = xv.z - (float)rb.z; q.w = xv.w - (float)rb.w;
        *(f32x4*)(outq + tok * DIMS + d0) = q;
    }
}

// ---------------------------------------------------------------------------
// Final deterministic reduction of per-block commit partials.
// ---------------------------------------------------------------------------
__global__ __launch_bounds__(1024)
void rvq_reduce(const float* __restrict__ partial, int n,
                float* __restrict__ outloss, float scale)
{
    __shared__ float s[1024];
    const int t = threadIdx.x;
    float v = 0.f;
    for (int i = t; i < n; i += 1024) v += partial[i];
    s[t] = v;
    __syncthreads();
    for (int o = 512; o > 0; o >>= 1) {
        if (t < o) s[t] += s[t + o];
        __syncthreads();
    }
    if (t == 0) *outloss = s[0] * scale;
}

// ---------------------------------------------------------------------------
extern "C" void kernel_launch(void* const* d_in, const int* in_sizes, int n_in,
                              void* d_out, int out_size, void* d_ws, size_t ws_size,
                              hipStream_t stream)
{
    (void)n_in; (void)out_size; (void)ws_size;

    const float* x = (const float*)d_in[0];
    const float* cb[8];
    int K[8];
    for (int i = 0; i < 8; ++i) {
        cb[i] = (const float*)d_in[1 + i];
        K[i]  = in_sizes[1 + i] / DIMS;
    }
    static const int Kpad[8]   = {16, 16, 16, 16, 32, 64, 128, 256};
    static const int rowoff[8] = {0, 16, 32, 48, 64, 96, 160, 288};
    // total padded rows = 544

    char* ws = (char*)d_ws;
    unsigned short* cbw = (unsigned short*)ws;                 // 544*768*2 = 835584 B
    float* partial      = (float*)(ws + 835584);               // up to 4096 B
    float* cnorm        = (float*)(ws + 835584 + 4096);        // 544*4 B

    const long ND   = (long)in_sizes[0];       // 32*2048*768
    const int  ntok = (int)(ND / DIMS);        // 65536
    const int  nblk = ntok / TILE;             // 512

    float* outq    = (float*)d_out;
    float* outidx  = outq + ND;
    float* outloss = outidx + (long)ntok * 8;

    for (int l = 0; l < 8; ++l)
        rvq_prep<<<Kpad[l], 256, 0, stream>>>(
            cb[l], K[l], cbw + (size_t)rowoff[l] * DIMS, cnorm + rowoff[l]);

    rvq_main<<<nblk, BLOCKDIM, LDS_BYTES, stream>>>(
        x, cb[0], cb[1], cb[2], cb[3], cb[4], cb[5], cb[6], cb[7],
        cbw, cnorm, outq, outidx, partial);

    rvq_reduce<<<1, 1024, 0, stream>>>(partial, nblk, outloss,
                                       0.25f / (float)ND);
}
